// WindowAttention_2499670966869
// MI455X (gfx1250) — matrix operand order repaired and checked by reading, not compiled, not run on hardware
//
#include <hip/hip_runtime.h>
#include <hip/hip_bf16.h>

typedef __attribute__((ext_vector_type(16))) _Float16 v16h;
typedef __attribute__((ext_vector_type(8)))  float    v8f;
typedef unsigned int uint32;

#define NTOK      49      // 7x7 window
#define DIMC      384
#define HEADS     12
#define HDIM      32
#define QKV_COLS  1152    // 3*DIM
#define SCALEF    0.17677669529663687f  // 32^-0.5

// ---------------------------------------------------------------------------
// WMMA fragment helpers (wave32, v_wmma_f32_16x16x32_f16)
// A: 16x32 f16, row-major source src[m*lda + k]
// ---------------------------------------------------------------------------
__device__ __forceinline__ v16h load_a_frag(const _Float16* src, int lda) {
  const int lane = threadIdx.x & 31;
  const int m    = lane & 15;
  const int half = lane >> 4;
  v16h a = {};
#pragma unroll
  for (int v = 0; v < 8; ++v) {
    const int k = ((v >> 2) << 4) + (half << 3) + ((v & 3) << 1);
    a[2 * v]     = src[m * lda + k];
    a[2 * v + 1] = src[m * lda + k + 1];
  }
  return a;
}

// B: 32x16 (KxN) sourced from a row-major N x K buffer (i.e. B = srcT),
// src[n*ldn + k] — contiguous f16 pairs along k -> 32/128-bit loads.
__device__ __forceinline__ v16h load_b_cm(const _Float16* src, int ldn) {
  const int lane = threadIdx.x & 31;
  const int n    = lane & 15;
  const int half = lane >> 4;
  v16h b = {};
#pragma unroll
  for (int v = 0; v < 8; ++v) {
    const int k = ((v >> 2) << 4) + (half << 3) + ((v & 3) << 1);
    b[2 * v]     = src[n * ldn + k];
    b[2 * v + 1] = src[n * ldn + k + 1];
  }
  return b;
}

// ---------------------------------------------------------------------------
// f32 -> f16 weight conversion (one-shot, tiny)
// ---------------------------------------------------------------------------
__global__ void cvt_f32_f16(const float* __restrict__ in,
                            _Float16* __restrict__ out, int n) {
  int i = blockIdx.x * blockDim.x + threadIdx.x;
  if (i < n) out[i] = (_Float16)in[i];
}

// ---------------------------------------------------------------------------
// QKV GEMM: C[M=100352, N=1152] = X[M,384] @ Wqkv^T + bqkv, f16 out.
// Block = 256 threads (8 waves), handles 32 rows (2 M-tiles) x 128 cols.
// Each wave computes two 16x16 tiles sharing one B fragment (B reuse, 24 WMMAs).
// ---------------------------------------------------------------------------
__global__ void qkv_gemm(const float* __restrict__ x,
                         const _Float16* __restrict__ wh,   // [1152][384] f16
                         const float* __restrict__ bias,    // [1152]
                         _Float16* __restrict__ qkvh) {     // [M][1152] f16
  __shared__ _Float16 As[32 * DIMC];                        // 24 KB
  const int mtile = blockIdx.x * 2;
  const int wave  = threadIdx.x >> 5;
  const int ntile = blockIdx.y * 8 + wave;

  // stage 32-row A tile, converting f32 -> f16 (coalesced over columns)
  for (int idx = threadIdx.x; idx < 32 * DIMC; idx += blockDim.x) {
    const int row = idx / DIMC, col = idx % DIMC;
    As[idx] = (_Float16)x[(size_t)(mtile * 16 + row) * DIMC + col];
  }
  __syncthreads();

  v8f acc0 = {}, acc1 = {};
#pragma unroll
  for (int kk = 0; kk < DIMC / 32; ++kk) {   // 12 K-steps, 24 WMMAs
    v16h b  = load_b_cm(&wh[(size_t)(ntile * 16) * DIMC + kk * 32], DIMC);
    v16h a0 = load_a_frag(&As[kk * 32], DIMC);
    v16h a1 = load_a_frag(&As[16 * DIMC + kk * 32], DIMC);
    acc0 = __builtin_amdgcn_wmma_f32_16x16x32_f16(false, a0, false, b,
                                                  (short)0, acc0, false, false);
    acc1 = __builtin_amdgcn_wmma_f32_16x16x32_f16(false, a1, false, b,
                                                  (short)0, acc1, false, false);
  }

  const int lane = threadIdx.x & 31;
  const int n = lane & 15, mb = (lane >> 4) * 8;
  const int ncol = ntile * 16 + n;
  const float bv = bias[ncol];
#pragma unroll
  for (int r = 0; r < 8; ++r) {
    const size_t row0 = (size_t)mtile * 16 + mb + r;
    qkvh[row0 * QKV_COLS + ncol]        = (_Float16)(acc0[r] + bv);
    qkvh[(row0 + 16) * QKV_COLS + ncol] = (_Float16)(acc1[r] + bv);
  }
}

// ---------------------------------------------------------------------------
// Window attention per (window b, head h). Block = 128 threads (4 waves).
// S = q k^T * scale + rpb[relidx] + mask[b%nW]; softmax; O = P v.
// V is staged transposed (vsT[d][tok]) so P@V B-frags are contiguous in LDS.
// ---------------------------------------------------------------------------
__global__ void win_attn(const _Float16* __restrict__ qkvh,  // [M][1152]
                         const float* __restrict__ mask,     // [nW][49][49]
                         const float* __restrict__ rpb,      // [169][12]
                         _Float16* __restrict__ outh,        // [M][384]
                         int nW) {
  const int b = blockIdx.x;
  const int h = blockIdx.y;
  __shared__ _Float16 qs[64 * 32], ks[64 * 32];   // zero-padded, row-major
  __shared__ _Float16 vsT[32 * 64];               // transposed: [d][tok]
  __shared__ float    Ss[64 * 64];
  __shared__ _Float16 Ps[64 * 64];
  const int tid = threadIdx.x;

  // stage q/k row-major as 32-bit pairs; v transposed (rows >= 49 zero-padded)
  for (int idx = tid; idx < 64 * 16; idx += 128) {
    const int row = idx >> 4, pr = idx & 15;
    uint32 qv = 0, kv = 0, vv = 0;
    if (row < NTOK) {
      const size_t base = (size_t)(b * NTOK + row) * QKV_COLS + h * HDIM;
      const uint32* p = (const uint32*)(qkvh + base);
      qv = p[pr];
      kv = p[DIMC / 2 + pr];       // +384 f16
      vv = p[DIMC + pr];           // +768 f16
    }
    ((uint32*)qs)[idx] = qv;
    ((uint32*)ks)[idx] = kv;
    _Float16 two[2];
    *(uint32*)two = vv;
    vsT[(pr * 2) * 64 + row]     = two[0];
    vsT[(pr * 2 + 1) * 64 + row] = two[1];
  }
  __syncthreads();

  const int wave = tid >> 5;
  const int lane = tid & 31;
  const int nn = lane & 15, mb = (lane >> 4) * 8;

  // ---- S = q @ k^T (wave owns M-tile = wave; K=32 is one WMMA step)
  {
    v16h a = load_a_frag(&qs[wave * 16 * 32], 32);
#pragma unroll
    for (int nt = 0; nt < 4; ++nt) {
      v8f acc = {};
      v16h bf = load_b_cm(&ks[nt * 16 * 32], 32);  // B = k^T
      acc = __builtin_amdgcn_wmma_f32_16x16x32_f16(false, a, false, bf,
                                                   (short)0, acc, false, false);
      const int j = nt * 16 + nn;
#pragma unroll
      for (int r = 0; r < 8; ++r) {
        const int i = wave * 16 + mb + r;
        float val = 0.0f;
        if (i < NTOK && j < NTOK) {
          const int ih = i / 7, iw = i % 7, jh = j / 7, jw = j % 7;
          const int ridx = (ih - jh + 6) * 13 + (iw - jw + 6);
          val = acc[r] * SCALEF + rpb[ridx * HEADS + h] +
                mask[(size_t)(b % nW) * NTOK * NTOK + i * NTOK + j];
        }
        Ss[i * 64 + j] = val;
      }
    }
  }
  __syncthreads();

  // ---- softmax rows (valid cols j<49); build f16 P with zero padding
  if (tid < 64) {
    const int i = tid;
    if (i < NTOK) {
      float m = -3.4e38f;
      for (int j = 0; j < NTOK; ++j) m = fmaxf(m, Ss[i * 64 + j]);
      float s = 0.f;
      for (int j = 0; j < NTOK; ++j) {
        float e = __expf(Ss[i * 64 + j] - m);
        Ss[i * 64 + j] = e;
        s += e;
      }
      const float inv = 1.0f / s;
      for (int j = 0; j < NTOK; ++j) Ps[i * 64 + j] = (_Float16)(Ss[i * 64 + j] * inv);
      for (int j = NTOK; j < 64; ++j) Ps[i * 64 + j] = (_Float16)0.f;
    } else {
      for (int j = 0; j < 64; ++j) Ps[i * 64 + j] = (_Float16)0.f;
    }
  }
  __syncthreads();

  // ---- O = P @ V : M=64 (wave-tile), N=32 (2 tiles), K=64 (2 steps)
#pragma unroll
  for (int nt = 0; nt < 2; ++nt) {
    v8f acc = {};
#pragma unroll
    for (int kt = 0; kt < 2; ++kt) {
      v16h a  = load_a_frag(&Ps[wave * 16 * 64 + kt * 32], 64);
      // B[k=tok][n=d] from transposed V: vsT[(nt*16+n)*64 + kt*32 + k]
      v16h bf = load_b_cm(&vsT[(nt * 16) * 64 + kt * 32], 64);
      acc = __builtin_amdgcn_wmma_f32_16x16x32_f16(false, a, false, bf,
                                                   (short)0, acc, false, false);
    }
    const int d = nt * 16 + nn;
#pragma unroll
    for (int r = 0; r < 8; ++r) {
      const int i = wave * 16 + mb + r;
      if (i < NTOK)
        outh[(size_t)(b * NTOK + i) * DIMC + h * HDIM + d] = (_Float16)acc[r];
    }
  }
}

// ---------------------------------------------------------------------------
// Output projection: out[M,384] = Ah[M,384] @ Wproj^T + bproj (f32 out)
// Same 2-M-tile / B-reuse scheme as qkv_gemm.
// ---------------------------------------------------------------------------
__global__ void proj_gemm(const _Float16* __restrict__ xh,   // [M][384] f16
                          const _Float16* __restrict__ wh,   // [384][384] f16
                          const float* __restrict__ bias,    // [384]
                          float* __restrict__ out) {         // [M][384] f32
  __shared__ _Float16 As[32 * DIMC];
  const int mtile = blockIdx.x * 2;
  const int wave  = threadIdx.x >> 5;
  const int ntile = blockIdx.y * 8 + wave;

  // stage A tile (already f16): copy as 32-bit words
  for (int idx = threadIdx.x; idx < 32 * (DIMC / 2); idx += blockDim.x) {
    const int row = idx / (DIMC / 2), pr = idx % (DIMC / 2);
    const uint32* p = (const uint32*)(xh + (size_t)(mtile * 16 + row) * DIMC);
    ((uint32*)As)[idx] = p[pr];
  }
  __syncthreads();

  v8f acc0 = {}, acc1 = {};
#pragma unroll
  for (int kk = 0; kk < DIMC / 32; ++kk) {   // 12 K-steps, 24 WMMAs
    v16h b  = load_b_cm(&wh[(size_t)(ntile * 16) * DIMC + kk * 32], DIMC);
    v16h a0 = load_a_frag(&As[kk * 32], DIMC);
    v16h a1 = load_a_frag(&As[16 * DIMC + kk * 32], DIMC);
    acc0 = __builtin_amdgcn_wmma_f32_16x16x32_f16(false, a0, false, b,
                                                  (short)0, acc0, false, false);
    acc1 = __builtin_amdgcn_wmma_f32_16x16x32_f16(false, a1, false, b,
                                                  (short)0, acc1, false, false);
  }

  const int lane = threadIdx.x & 31;
  const int n = lane & 15, mb = (lane >> 4) * 8;
  const int ncol = ntile * 16 + n;
  const float bv = bias[ncol];
#pragma unroll
  for (int r = 0; r < 8; ++r) {
    const size_t row0 = (size_t)mtile * 16 + mb + r;
    out[row0 * DIMC + ncol]        = acc0[r] + bv;
    out[(row0 + 16) * DIMC + ncol] = acc1[r] + bv;
  }
}

// ---------------------------------------------------------------------------
extern "C" void kernel_launch(void* const* d_in, const int* in_sizes, int n_in,
                              void* d_out, int out_size, void* d_ws, size_t ws_size,
                              hipStream_t stream) {
  const float* x      = (const float*)d_in[0];   // (2048, 49, 384)
  const float* mask   = (const float*)d_in[1];   // (64, 49, 49)
  const float* qkv_w  = (const float*)d_in[2];   // (1152, 384)
  const float* qkv_b  = (const float*)d_in[3];   // (1152,)
  const float* proj_w = (const float*)d_in[4];   // (384, 384)
  const float* proj_b = (const float*)d_in[5];   // (384,)
  const float* rpb    = (const float*)d_in[6];   // (169, 12)
  float* out = (float*)d_out;

  const int Bwin   = in_sizes[0] / (NTOK * DIMC);         // 2048
  const int nW     = in_sizes[1] / (NTOK * NTOK);         // 64
  const int Mrows  = Bwin * NTOK;                         // 100352 (divisible by 32)
  const int Mtiles = Mrows / 16;                          // 6272 (even)

  // workspace carve-out (~310 MB total, all f16 intermediates)
  char* ws = (char*)d_ws;
  _Float16* qkvwh  = (_Float16*)ws; ws += sizeof(_Float16) * (size_t)QKV_COLS * DIMC;
  _Float16* projwh = (_Float16*)ws; ws += sizeof(_Float16) * (size_t)DIMC * DIMC;
  _Float16* qkvh   = (_Float16*)ws; ws += sizeof(_Float16) * (size_t)Mrows * QKV_COLS;
  _Float16* aouth  = (_Float16*)ws; ws += sizeof(_Float16) * (size_t)Mrows * DIMC;
  (void)ws_size; (void)n_in; (void)out_size;

  cvt_f32_f16<<<(QKV_COLS * DIMC + 255) / 256, 256, 0, stream>>>(qkv_w, qkvwh,
                                                                 QKV_COLS * DIMC);
  cvt_f32_f16<<<(DIMC * DIMC + 255) / 256, 256, 0, stream>>>(proj_w, projwh,
                                                             DIMC * DIMC);
  qkv_gemm<<<dim3(Mtiles / 2, QKV_COLS / 128), 256, 0, stream>>>(x, qkvwh, qkv_b,
                                                                 qkvh);
  win_attn<<<dim3(Bwin, HEADS), 128, 0, stream>>>(qkvh, mask, rpb, aouth, nW);
  proj_gemm<<<dim3(Mtiles / 2, DIMC / 128), 256, 0, stream>>>(aouth, projwh,
                                                              proj_b, out);
}
